// MultiHeadAttention_46136538693743
// MI455X (gfx1250) — matrix operand order repaired and checked by reading, not compiled, not run on hardware
//
#include <hip/hip_runtime.h>
#include <hip/hip_bf16.h>
#include <stdint.h>

// ---------------------------------------------------------------------------
// Multi-head attention for MI455X (gfx1250), wave32 + V_WMMA_F32_16X16X32_BF16
// + GLOBAL_LOAD_ASYNC_TO_LDS staging (ASYNCcnt) in the flash-attention core.
// B=2, S=2048, D=1024, H=16, Hd=64.
// ---------------------------------------------------------------------------

typedef __attribute__((ext_vector_type(16))) __bf16 v16bf;
typedef __attribute__((ext_vector_type(8)))  __bf16 v8bf;
typedef __attribute__((ext_vector_type(8)))  float  v8f;

#define BATCH   2
#define SEQ     2048
#define EMBED   1024
#define HEADS   16
#define HDIM    64
#define ROWS    (BATCH * SEQ)      // 4096

__device__ __forceinline__ v8f v8f_zero() {
  v8f z;
#pragma unroll
  for (int i = 0; i < 8; ++i) z[i] = 0.0f;
  return z;
}

__device__ __forceinline__ v8f wmma_bf16(v16bf a, v16bf b, v8f c) {
  // (neg_a, A, neg_b, B, c_mod, C, reuse_a, reuse_b)
  return __builtin_amdgcn_wmma_f32_16x16x32_bf16(false, a, false, b, (short)0, c,
                                                 false, false);
}

// Async copy 16 bytes global -> LDS (per lane), tracked by ASYNCcnt.
// Flat LDS addresses carry the LDS byte offset in bits [31:0] (ISA 10.2),
// so the low 32 bits of a generic pointer to __shared__ are the ds offset.
__device__ __forceinline__ void async_ld_b128(uint32_t lds_off, const void* g) {
  asm volatile("global_load_async_to_lds_b128 %0, %1, off"
               :: "v"(lds_off), "v"(g) : "memory");
}
__device__ __forceinline__ void wait_async0() {
  asm volatile("s_wait_asynccnt 0" ::: "memory");
}
__device__ __forceinline__ uint32_t lds_off32(const void* p) {
  return (uint32_t)(uintptr_t)p;
}

// A-operand fragment (16x32, MxK). Per ISA: lanes 0-15 rows M=lane,
// elements 0..7 -> K = kb..kb+7, elements 8..15 -> K = 16+kb..16+kb+7,
// kb = 8 for lanes 16-31. `tile` points at row 0 of a 16-row tile,
// row-major with `stride` elements per row (K contiguous).
__device__ __forceinline__ v16bf frag_ld_a(const __bf16* tile, int stride, int k0) {
  int lane = threadIdx.x & 31;
  int r  = lane & 15;
  int kb = (lane >> 4) << 3;        // 0 or 8
  const __bf16* p = tile + (size_t)r * stride + k0 + kb;
  v8bf lo = *(const v8bf*)(p);
  v8bf hi = *(const v8bf*)(p + 16);
  v16bf f;
#pragma unroll
  for (int i = 0; i < 8; ++i) { f[i] = lo[i]; f[8 + i] = hi[i]; }
  return f;
}

// B-operand fragment (32x16, KxN). Same K order as A: lanes 0-15 hold K=0..7,16..23,
// lanes 16-31 hold K=8..15,24..31, n = lane&15. We store B k-major: `tile` points at
// a 16-"row" tile where row index = n and columns = K (stride elems per n).
__device__ __forceinline__ v16bf frag_ld_b(const __bf16* tile, int stride, int k0) {
  int lane = threadIdx.x & 31;
  int n  = lane & 15;
  int kb = (lane >> 4) << 3;        // 0 or 8
  const __bf16* p = tile + (size_t)n * stride + k0 + kb;
  v8bf lo = *(const v8bf*)(p);
  v8bf hi = *(const v8bf*)(p + 16);
  v16bf f;
#pragma unroll
  for (int i = 0; i < 8; ++i) { f[i] = lo[i]; f[8 + i] = hi[i]; }
  return f;
}

// ---------------------------------------------------------------------------
// Kernel 1: fp32 -> bf16 copy (x)
// ---------------------------------------------------------------------------
__global__ __launch_bounds__(256) void cvt_bf16_kernel(const float* __restrict__ in,
                                                       __bf16* __restrict__ out, int n) {
  int i = blockIdx.x * blockDim.x + threadIdx.x;
  if (i < n) out[i] = (__bf16)in[i];
}

// ---------------------------------------------------------------------------
// Kernel 2: fp32 [K=1024][N=1024] -> bf16 transposed [N][K]
// ---------------------------------------------------------------------------
__global__ __launch_bounds__(256) void transpose_bf16_kernel(const float* __restrict__ in,
                                                             __bf16* __restrict__ out) {
  int i = blockIdx.x * blockDim.x + threadIdx.x;   // over 1024*1024
  int n = i >> 10;
  int k = i & 1023;
  out[(size_t)n * 1024 + k] = (__bf16)in[(size_t)k * 1024 + n];
}

// ---------------------------------------------------------------------------
// Kernel 3: GEMM  C[4096,1024] = A(bf16,[4096,1024]) @ Bt(bf16,[1024 n][1024 k])^T
// + bias, with output-layout modes:
//   mode 0: bf16, Q/K layout  [B,H,S,Hd]
//   mode 1: bf16, Vt layout   [B,H,Hd,S]
//   mode 2: f32,  row-major   [4096,1024]  (final output)
// 8 waves/block, each wave computes a 32x64 tile (8 WMMA accumulators).
// L2 (192MB) holds the full operand set; fragments load straight from global.
// ---------------------------------------------------------------------------
__global__ __launch_bounds__(256) void gemm_bf16_kernel(const __bf16* __restrict__ A,
                                                        const __bf16* __restrict__ Bt,
                                                        const float* __restrict__ bias,
                                                        void* __restrict__ outp,
                                                        int mode) {
  const int wid = (blockIdx.x * blockDim.x + threadIdx.x) >> 5; // global wave id
  const int wm = wid >> 4;          // 0..127  (M tiles of 32)
  const int wn = wid & 15;          // 0..15   (N tiles of 64)
  const int rowbase = wm * 32;
  const int colbase = wn * 64;

  const int lane = threadIdx.x & 31;
  const int hi = lane >> 4;
  const int lr = lane & 15;

  v8f acc[2][4];
#pragma unroll
  for (int mt = 0; mt < 2; ++mt)
#pragma unroll
    for (int nt = 0; nt < 4; ++nt) acc[mt][nt] = v8f_zero();

  const __bf16* A0 = A + (size_t)rowbase * EMBED;
  const __bf16* A1 = A0 + (size_t)16 * EMBED;
  const __bf16* B0 = Bt + (size_t)colbase * EMBED;

  for (int k0 = 0; k0 < EMBED; k0 += 32) {
    v16bf a0 = frag_ld_a(A0, EMBED, k0);
    v16bf a1 = frag_ld_a(A1, EMBED, k0);
#pragma unroll
    for (int nt = 0; nt < 4; ++nt) {
      v16bf b = frag_ld_b(B0 + (size_t)(nt * 16) * EMBED, EMBED, k0);
      acc[0][nt] = wmma_bf16(a0, b, acc[0][nt]);
      acc[1][nt] = wmma_bf16(a1, b, acc[1][nt]);
    }
  }

#pragma unroll
  for (int mt = 0; mt < 2; ++mt) {
#pragma unroll
    for (int nt = 0; nt < 4; ++nt) {
      int c = colbase + nt * 16 + lr;
      float bs = bias[c];
#pragma unroll
      for (int r = 0; r < 8; ++r) {
        int rg = rowbase + mt * 16 + r + 8 * hi;   // global row in [0,4096)
        float v = acc[mt][nt][r] + bs;
        if (mode == 2) {
          ((float*)outp)[(size_t)rg * EMBED + c] = v;
        } else {
          int b = rg >> 11;          // batch
          int s = rg & 2047;         // seq
          int h = c >> 6;            // head
          int d = c & 63;            // head dim
          __bf16* O = (__bf16*)outp;
          if (mode == 0) {  // [B,H,S,Hd]
            O[(((size_t)(b * HEADS + h) * SEQ) + s) * HDIM + d] = (__bf16)v;
          } else {          // [B,H,Hd,S]
            O[(((size_t)(b * HEADS + h) * HDIM) + d) * SEQ + s] = (__bf16)v;
          }
        }
      }
    }
  }
}

// ---------------------------------------------------------------------------
// Kernel 4: flash attention.  One block per (b,h, 64-row query block).
// 4 waves x 16 query rows.  Online softmax; K and V tiles are staged into LDS
// once per block with GLOBAL_LOAD_ASYNC_TO_LDS_B128 (ASYNCcnt) and shared by
// all 4 waves; P staged bf16 through per-wave LDS tiles.
// Q,K: [B,H,S,Hd] bf16; Vt: [B,H,Hd,S] bf16; ctx out: [B*S, 1024] bf16.
// ---------------------------------------------------------------------------
__global__ __launch_bounds__(128) void flash_attn_kernel(const __bf16* __restrict__ Q,
                                                         const __bf16* __restrict__ K,
                                                         const __bf16* __restrict__ Vt,
                                                         __bf16* __restrict__ ctx) {
  const int bh = blockIdx.x >> 5;     // b*16+h
  const int qb = blockIdx.x & 31;     // query block (64 rows)
  const int b  = bh >> 4;
  const int h  = bh & 15;

  const int tid  = threadIdx.x;
  const int wave = tid >> 5;
  const int lane = tid & 31;
  const int hi = lane >> 4;
  const int lr = lane & 15;

  const __bf16* Qh = Q  + (size_t)bh * SEQ * HDIM;
  const __bf16* Kh = K  + (size_t)bh * SEQ * HDIM;
  const __bf16* Vh = Vt + (size_t)bh * HDIM * SEQ;

  const int rowbase = qb * 64 + wave * 16;   // this wave's 16 query rows

  // Q fragments for the full Hd=64 (two K=32 fragments).
  v16bf qf0 = frag_ld_a(Qh + (size_t)rowbase * HDIM, HDIM, 0);
  v16bf qf1 = frag_ld_a(Qh + (size_t)rowbase * HDIM, HDIM, 32);

  __shared__ __bf16 Kt[64][64];      // key block  [key n][d k]   8 KB
  __shared__ __bf16 Vs[64][64];      // value blk  [d n][s k]     8 KB
  __shared__ __bf16 Pt[4][16][64];   // per-wave P tiles          8 KB

  const uint32_t kt_base = lds_off32(&Kt[0][0]);
  const uint32_t vs_base = lds_off32(&Vs[0][0]);

  float mrun[8], lrun[8];
  v8f o[4];
#pragma unroll
  for (int r = 0; r < 8; ++r) { mrun[r] = -3.0e38f; lrun[r] = 0.0f; }
#pragma unroll
  for (int nt = 0; nt < 4; ++nt) o[nt] = v8f_zero();

  for (int j = 0; j < SEQ / 64; ++j) {
    // ---- async-stage K tile (contiguous 8 KB) and V tile (64 x 128B rows) ----
    {
      const char* kg = (const char*)(Kh + (size_t)j * 64 * HDIM);
#pragma unroll
      for (int c = 0; c < 4; ++c) {
        uint32_t off = (uint32_t)(tid * 16 + c * 2048);
        async_ld_b128(kt_base + off, kg + off);
      }
      const char* vg = (const char*)(Vh + j * 64);
#pragma unroll
      for (int c = 0; c < 4; ++c) {
        int i   = c * 128 + tid;     // 16B-chunk index 0..511
        int row = i >> 3;            // d row 0..63
        int col = i & 7;             // 16B chunk within 128B row
        async_ld_b128(vs_base + (uint32_t)(row * 128 + col * 16),
                      vg + (size_t)row * SEQ * 2 + col * 16);
      }
      wait_async0();
      __syncthreads();
    }

    // ---- S = (Q K^T) * 1/sqrt(Hd), B operand from LDS K tile ----
    v8f s[4];
#pragma unroll
    for (int nt = 0; nt < 4; ++nt) {
      s[nt] = v8f_zero();
      v16bf b0 = frag_ld_b(&Kt[nt * 16][0], 64, 0);
      s[nt] = wmma_bf16(qf0, b0, s[nt]);
      v16bf b1 = frag_ld_b(&Kt[nt * 16][0], 64, 32);
      s[nt] = wmma_bf16(qf1, b1, s[nt]);
    }
#pragma unroll
    for (int nt = 0; nt < 4; ++nt)
#pragma unroll
      for (int r = 0; r < 8; ++r) s[nt][r] *= 0.125f;   // 1/sqrt(64)

    // row max across 4 col-tiles then across the 16 lanes of this half-wave
    float rm[8];
#pragma unroll
    for (int r = 0; r < 8; ++r) {
      rm[r] = fmaxf(fmaxf(s[0][r], s[1][r]), fmaxf(s[2][r], s[3][r]));
#pragma unroll
      for (int m = 1; m < 16; m <<= 1) rm[r] = fmaxf(rm[r], __shfl_xor(rm[r], m, 32));
    }

    float corr[8], rs[8];
#pragma unroll
    for (int r = 0; r < 8; ++r) {
      float mnew = fmaxf(mrun[r], rm[r]);
      corr[r] = __expf(mrun[r] - mnew);
      mrun[r] = mnew;
      rs[r] = 0.0f;
    }
#pragma unroll
    for (int nt = 0; nt < 4; ++nt)
#pragma unroll
      for (int r = 0; r < 8; ++r) {
        float p = __expf(s[nt][r] - mrun[r]);
        s[nt][r] = p;
        rs[r] += p;
      }
#pragma unroll
    for (int r = 0; r < 8; ++r) {
#pragma unroll
      for (int m = 1; m < 16; m <<= 1) rs[r] += __shfl_xor(rs[r], m, 32);
      lrun[r] = lrun[r] * corr[r] + rs[r];
    }
#pragma unroll
    for (int nt = 0; nt < 4; ++nt)
#pragma unroll
      for (int r = 0; r < 8; ++r) o[nt][r] *= corr[r];

    // stage P (bf16) in this wave's private LDS tile
#pragma unroll
    for (int nt = 0; nt < 4; ++nt)
#pragma unroll
      for (int r = 0; r < 8; ++r)
        Pt[wave][r + 8 * hi][nt * 16 + lr] = (__bf16)s[nt][r];

    // ---- O += P (16x64) @ V_j (64x64), B operand from LDS V tile ----
    v16bf p0 = frag_ld_a(&Pt[wave][0][0], 64, 0);
    v16bf p1 = frag_ld_a(&Pt[wave][0][0], 64, 32);
#pragma unroll
    for (int nt = 0; nt < 4; ++nt) {
      v16bf b0 = frag_ld_b(&Vs[nt * 16][0], 64, 0);
      o[nt] = wmma_bf16(p0, b0, o[nt]);
      v16bf b1 = frag_ld_b(&Vs[nt * 16][0], 64, 32);
      o[nt] = wmma_bf16(p1, b1, o[nt]);
    }

    __syncthreads();   // protect K/V tiles before next iteration's overwrite
  }

  // normalize and write ctx [B*S, 1024] bf16
#pragma unroll
  for (int r = 0; r < 8; ++r) lrun[r] = 1.0f / lrun[r];
#pragma unroll
  for (int nt = 0; nt < 4; ++nt)
#pragma unroll
    for (int r = 0; r < 8; ++r) {
      int srow = rowbase + r + 8 * hi;                  // 0..2047
      int c = h * HDIM + nt * 16 + lr;                  // 0..1023
      ctx[((size_t)(b * SEQ + srow)) * EMBED + c] = (__bf16)(o[nt][r] * lrun[r]);
    }
}

// ---------------------------------------------------------------------------
// Host launcher
// ---------------------------------------------------------------------------
extern "C" void kernel_launch(void* const* d_in, const int* in_sizes, int n_in,
                              void* d_out, int out_size, void* d_ws, size_t ws_size,
                              hipStream_t stream) {
  const float* x  = (const float*)d_in[0];
  const float* wq = (const float*)d_in[1];
  const float* bq = (const float*)d_in[2];
  const float* wk = (const float*)d_in[3];
  const float* bk = (const float*)d_in[4];
  const float* wv = (const float*)d_in[5];
  const float* bv = (const float*)d_in[6];
  const float* wo = (const float*)d_in[7];
  const float* bo = (const float*)d_in[8];

  char* ws = (char*)d_ws;
  const size_t MB = 1024 * 1024;
  __bf16* xb   = (__bf16*)(ws + 0 * MB);    // 8 MiB  [4096,1024]
  __bf16* wqt  = (__bf16*)(ws + 8 * MB);    // 2 MiB  [N,K]
  __bf16* wkt  = (__bf16*)(ws + 10 * MB);
  __bf16* wvt  = (__bf16*)(ws + 12 * MB);
  __bf16* wot  = (__bf16*)(ws + 14 * MB);
  __bf16* Qb   = (__bf16*)(ws + 16 * MB);   // 8 MiB  [B,H,S,Hd]
  __bf16* Kb   = (__bf16*)(ws + 24 * MB);   // 8 MiB  [B,H,S,Hd]
  __bf16* Vtb  = (__bf16*)(ws + 32 * MB);   // 8 MiB  [B,H,Hd,S]
  __bf16* ctx  = (__bf16*)(ws + 40 * MB);   // 8 MiB  [4096,1024]

  // 1) x -> bf16
  cvt_bf16_kernel<<<(ROWS * EMBED) / 256, 256, 0, stream>>>(x, xb, ROWS * EMBED);

  // 2) weights -> transposed bf16 (k-major B operands)
  transpose_bf16_kernel<<<(EMBED * EMBED) / 256, 256, 0, stream>>>(wq, wqt);
  transpose_bf16_kernel<<<(EMBED * EMBED) / 256, 256, 0, stream>>>(wk, wkt);
  transpose_bf16_kernel<<<(EMBED * EMBED) / 256, 256, 0, stream>>>(wv, wvt);
  transpose_bf16_kernel<<<(EMBED * EMBED) / 256, 256, 0, stream>>>(wo, wot);

  // 3) projections: 2048 waves = 256 blocks x 8 waves
  gemm_bf16_kernel<<<256, 256, 0, stream>>>(xb, wqt, bq, (void*)Qb, 0);
  gemm_bf16_kernel<<<256, 256, 0, stream>>>(xb, wkt, bk, (void*)Kb, 0);
  gemm_bf16_kernel<<<256, 256, 0, stream>>>(xb, wvt, bv, (void*)Vtb, 1);

  // 4) flash attention: B*H*(S/64) = 1024 blocks x 4 waves
  flash_attn_kernel<<<1024, 128, 0, stream>>>(Qb, Kb, Vtb, ctx);

  // 5) output projection -> f32 d_out
  gemm_bf16_kernel<<<256, 256, 0, stream>>>(ctx, wot, bo, d_out, 2);
}
